// RTMCCHead_1563368096604
// MI455X (gfx1250) — matrix operand order repaired and checked by reading, not compiled, not run on hardware
//
#include <hip/hip_runtime.h>
#include <hip/hip_bf16.h>

typedef __attribute__((ext_vector_type(16))) _Float16 v16h;
typedef __attribute__((ext_vector_type(8)))  float    v8f;
typedef __attribute__((ext_vector_type(4)))  unsigned int v4u;
typedef __attribute__((ext_vector_type(8)))  int      v8i;
typedef __attribute__((ext_vector_type(4)))  int      v4i;
typedef _Float16 half_t;

#define NIMG 256
#define KP   133
#define ROWS (NIMG*KP)      // 34048 = 2128*16
#define CIN  1024
#define IH   12
#define IW   9
#define HW   108
#define KCONV (CIN*49)      // 50176
#define NPOS (NIMG*HW)      // 27648
#define MPADC 144
#define ED   512
#define DQK  128
#define QPAD 144
#define VPAD 160
#define DM   256
#define NX   576
#define NY   768

__constant__ int c_FLIP[KP] = {
  0, 2, 1, 4, 3, 6, 5, 8, 7, 10, 9, 12, 11, 14, 13, 16, 15, 20, 21, 22, 17,
  18, 19, 39, 38, 37, 36, 35, 34, 33, 32, 31, 30, 29, 28, 27, 26, 25, 24, 23,
  49, 48, 47, 46, 45, 44, 43, 42, 41, 40, 50, 51, 52, 53, 58, 57, 56, 55, 54,
  68, 67, 66, 65, 70, 69, 62, 61, 60, 59, 64, 63, 77, 76, 75, 74, 73, 72, 71,
  82, 81, 80, 79, 78, 87, 86, 85, 84, 83, 90, 89, 88, 112, 113, 114, 115, 116,
  117, 118, 119, 120, 121, 122, 123, 124, 125, 126, 127, 128, 129, 130, 131,
  132, 91, 92, 93, 94, 95, 96, 97, 98, 99, 100, 101, 102, 103, 104, 105, 106,
  107, 108, 109, 110, 111 };

// ---------------- WMMA fragment helpers (16x16x32 f16 -> f32) ----------------

__device__ __forceinline__ v8f wmma16(v16h a, v16h b, v8f c) {
  return __builtin_amdgcn_wmma_f32_16x16x32_f16(false, a, false, b, (short)0,
                                                c, false, false);
}

// A fragment: row-major M x K, tile (m0,k0).  Contiguous K layout:
// lane L: M = L%16; VGPR v holds K = {16h+2v, 16h+2v+1}
__device__ __forceinline__ v16h load_a16(const half_t* __restrict__ A, long lda,
                                         int m0, int k0, int lane) {
  int h = lane >> 4;
  const half_t* p = A + (long)(m0 + (lane & 15)) * lda + k0;
  v16h a;
#pragma unroll
  for (int v = 0; v < 8; ++v) {
    int kb = 16 * h + 2 * v;
    a[2 * v] = p[kb];
    a[2 * v + 1] = p[kb + 1];
  }
  return a;
}

// B fragment from row-major weight W[N][K]  (B = W^T): lane n = n0+L%16,
// element e -> K = k0 + 16h + e  (contiguous along K -> vectorizable load)
__device__ __forceinline__ v16h load_bT16(const half_t* __restrict__ W, long ldw,
                                          int k0, int n0, int lane) {
  int h = lane >> 4;
  const half_t* p = W + (long)(n0 + (lane & 15)) * ldw + (k0 + 16 * h);
  v16h b;
#pragma unroll
  for (int e = 0; e < 16; ++e) b[e] = p[e];
  return b;
}

// B fragment from row-major K x N matrix
__device__ __forceinline__ v16h load_brm16(const half_t* __restrict__ B, long ldb,
                                           int k0, int n0, int lane) {
  int h = lane >> 4;
  const half_t* p = B + (long)(k0 + 16 * h) * ldb + (n0 + (lane & 15));
  v16h b;
#pragma unroll
  for (int e = 0; e < 16; ++e) b[e] = p[e * ldb];
  return b;
}

template <int KS>
__device__ __forceinline__ v8f gemm_bT(const half_t* A, long lda,
                                       const half_t* W, long ldw,
                                       int m0, int n0, int lane) {
  v8f acc = {};
#pragma unroll
  for (int ks = 0; ks < KS; ++ks)
    acc = wmma16(load_a16(A, lda, m0, ks * 32, lane),
                 load_bT16(W, ldw, ks * 32, n0, lane), acc);
  return acc;
}

template <int KS>
__device__ __forceinline__ v8f gemm_brm(const half_t* A, long lda,
                                        const half_t* B, long ldb,
                                        int m0, int n0, int lane) {
  v8f acc = {};
#pragma unroll
  for (int ks = 0; ks < KS; ++ks)
    acc = wmma16(load_a16(A, lda, m0, ks * 32, lane),
                 load_brm16(B, ldb, ks * 32, n0, lane), acc);
  return acc;
}

// ---------------- prepack / convert ----------------

// dst[n][k] (f16, Nd x Kd) = src[n][k] (f32, Ns x Ks), zero-padded
__global__ void k_cvt_pad(half_t* __restrict__ dst, const float* __restrict__ src,
                          int Nd, int Ns, int Kd, int Ks) {
  long i = blockIdx.x * (long)blockDim.x + threadIdx.x;
  long tot = (long)Nd * Kd;
  if (i >= tot) return;
  int n = (int)(i / Kd), k = (int)(i % Kd);
  float v = (n < Ns && k < Ks) ? src[(long)n * Ks + k] : 0.f;
  dst[i] = (half_t)v;
}

// ---------------- conv: implicit GEMM, M=co(144) N=pos(27648) K=50176 --------
// Block = 96 threads (3 waves).  Waves split over M (3 x 16 rows), share one
// 16-column N tile.  Per K-step:
//   * wave 0 issues a Tensor-Data-Mover 2D tile load (48 x 32 f16 of the
//     prepacked weight matrix) into LDS  -> tensor_load_to_lds / TENSORcnt
//   * all 96 threads cooperatively gather the 32 x 16 im2col B tile into LDS
//   * barrier, fragments read back through DS, one WMMA per wave.

__global__ void __launch_bounds__(96)
k_conv(const half_t* __restrict__ Aw, const half_t* __restrict__ fin,
       const float* __restrict__ bias, float* __restrict__ feats0) {
  __shared__ half_t sA[48 * 32];   // A tile (TDM destination)
  __shared__ half_t sB[32 * 16];   // im2col B tile
  __shared__ int    sNim[16], sOh[16], sOw[16];

  int tid = threadIdx.x;
  int lane = tid & 31;
  int wave = tid >> 5;
  int h = lane >> 4;
  int nlane = lane & 15;
  int m0b = blockIdx.y * 48;            // block's first output-channel row
  int m0 = m0b + wave * 16;             // this wave's M tile
  int pos0 = blockIdx.x * 16;           // shared N tile

  if (tid < 16) {
    int pos = pos0 + tid;
    int nimg = pos / HW;
    int rem = pos % HW;
    sNim[tid] = nimg * CIN;
    sOh[tid] = rem / IW;
    sOw[tid] = rem % IW;
  }

  // --- TDM descriptor (D#) for a 48-row x 32-col f16 tile of Aw -------------
  unsigned ldsA = (unsigned)(unsigned long long)(uintptr_t)&sA[0]; // LDS byte offset
  unsigned long long gbase =
      (unsigned long long)(uintptr_t)Aw + (unsigned long long)m0b * KCONV * 2u;

  v8i g1 = { (int)(1u << 16),                         // data_size = 1 (2 bytes)
             (int)((KCONV & 0xFFFF) << 16),           // tensor_dim0 [63:48]
             (int)((KCONV >> 16) | ((MPADC & 0xFFFF) << 16)), // dim0 hi, dim1 lo
             (int)(32u << 16),                        // tile_dim0 = 32
             48,                                      // tile_dim1 = 48
             (int)KCONV,                              // tensor_dim0_stride lo32
             0, 0 };
  v4i gz = { 0, 0, 0, 0 };

  v8f acc = {};
  for (int k0 = 0; k0 < KCONV; k0 += 32) {
    __syncthreads();  // previous-step LDS reads complete
    if (wave == 0) {
      unsigned long long ga = gbase + (unsigned)k0 * 2u;
      v4u g0 = { 1u,                                   // count = 1 (valid D#)
                 ldsA,                                 // lds_addr
                 (unsigned)(ga & 0xFFFFFFFFu),
                 (unsigned)((ga >> 32) & 0x01FFFFFFu) | (2u << 30) }; // type=2
#if defined(__clang_major__) && (__clang_major__ >= 23)
      v8i g4 = { 0, 0, 0, 0, 0, 0, 0, 0 };
      __builtin_amdgcn_tensor_load_to_lds(g0, g1, gz, gz, g4, 0);
#else
      __builtin_amdgcn_tensor_load_to_lds(g0, g1, gz, gz, 0);
#endif
    }
    // cooperative branch-free im2col gather of the 32x16 B tile
    for (int i = tid; i < 512; i += 96) {
      int kr = i >> 4;
      int nn = i & 15;
      int k = k0 + kr;
      int ci = k / 49;
      int r = k - ci * 49;
      int kh = r / 7;
      int kw = r - kh * 7;
      int ihh = sOh[nn] + kh - 3;
      int iww = sOw[nn] + kw - 3;
      bool ok = ((unsigned)ihh < (unsigned)IH) & ((unsigned)iww < (unsigned)IW);
      long idx = ok ? (((long)(sNim[nn] + ci) * IH + ihh) * IW + iww) : 0;
      __builtin_prefetch(fin + idx + 4 * IH * IW, 0, 1);  // global_prefetch_b8
      half_t raw = fin[idx];
      sB[i] = ok ? raw : (half_t)0.f;
    }
    __builtin_amdgcn_s_wait_tensorcnt(0);  // A tile landed in LDS
    __syncthreads();                       // B tile + A tile visible to block

    v16h a;
#pragma unroll
    for (int v = 0; v < 8; ++v) {
      int kb = 16 * h + 2 * v;
      a[2 * v] = sA[(wave * 16 + nlane) * 32 + kb];
      a[2 * v + 1] = sA[(wave * 16 + nlane) * 32 + kb + 1];
    }
    v16h b;
#pragma unroll
    for (int e = 0; e < 16; ++e) b[e] = sB[(16 * h + e) * 16 + nlane];
    acc = wmma16(a, b, acc);
  }

  int nimgc = sNim[nlane] / CIN;
  int rem = sOh[nlane] * IW + sOw[nlane];
#pragma unroll
  for (int v = 0; v < 8; ++v) {
    int co = m0 + v + 8 * h;
    if (co < KP)
      feats0[((long)nimgc * KP + co) * HW + rem] = acc[v] + bias[co];
  }
}

// ---------------- RMS scale-norm: x / max(rms, eps) * g -> f16 (K padded) ----

__global__ void k_norm(const float* __restrict__ src, half_t* __restrict__ dst,
                       int K, int Kd, const float* __restrict__ g) {
  int row = blockIdx.x * blockDim.x + threadIdx.x;
  if (row >= ROWS) return;
  const float* p = src + (long)row * K;
  float ss = 0.f;
  for (int i = 0; i < K; ++i) { float v = p[i]; ss += v * v; }
  float norm = sqrtf(ss / (float)K);
  float sc = g[0] / fmaxf(norm, 1e-5f);
  half_t* q = dst + (long)row * Kd;
  for (int i = 0; i < K; ++i) q[i] = (half_t)(p[i] * sc);
  for (int i = K; i < Kd; ++i) q[i] = (half_t)0.f;
}

// ---------------- MLP: (34048 x 128) @ mlp_w^T -> feats f32 (34048 x 256) ----

__global__ void k_gemm_mlp(const half_t* __restrict__ A, const half_t* __restrict__ W,
                           float* __restrict__ out) {
  int lane = threadIdx.x & 31, wave = threadIdx.x >> 5;
  int m0 = blockIdx.y * 16, n0 = (blockIdx.x * 4 + wave) * 16;
  v8f acc = gemm_bT<4>(A, 128, W, 128, m0, n0, lane);
  int h = lane >> 4, c = n0 + (lane & 15);
#pragma unroll
  for (int v = 0; v < 8; ++v)
    out[(long)(m0 + v + 8 * h) * DM + c] = acc[v];
}

// ---------------- UV: (34048 x 256) @ uv_w^T, SiLU, split u/v/q/k ------------

__global__ void k_gemm_uv(const half_t* __restrict__ A, const half_t* __restrict__ W,
                          half_t* __restrict__ u, half_t* __restrict__ vpn,
                          half_t* __restrict__ qpn, half_t* __restrict__ kpn,
                          const float* __restrict__ gamma, const float* __restrict__ beta) {
  int lane = threadIdx.x & 31, wave = threadIdx.x >> 5;
  int m0 = blockIdx.y * 16, n0 = (blockIdx.x * 4 + wave) * 16;
  v8f acc = gemm_bT<8>(A, 256, W, 256, m0, n0, lane);
  int h = lane >> 4, c = n0 + (lane & 15);
#pragma unroll
  for (int v = 0; v < 8; ++v) {
    int row = m0 + v + 8 * h;
    float x = acc[v];
    float s = x / (1.f + __expf(-x));     // SiLU
    if (c < ED) {
      u[(long)row * ED + c] = (half_t)s;
    } else if (c < 2 * ED) {
      int nimg = row / KP, kk = row % KP;
      vpn[((long)nimg * VPAD + kk) * ED + (c - ED)] = (half_t)s;
    } else {
      int nimg = row / KP, kk = row % KP;
      int cp = c - 2 * ED;
      qpn[((long)nimg * QPAD + kk) * DQK + cp] = (half_t)(s * gamma[cp] + beta[cp]);
      kpn[((long)nimg * QPAD + kk) * DQK + cp] = (half_t)(s * gamma[DQK + cp] + beta[DQK + cp]);
    }
  }
}

// ---------------- zero the padding rows/cols used by attention tiling --------

__global__ void k_padzero(half_t* __restrict__ vpn, half_t* __restrict__ qpn,
                          half_t* __restrict__ kpn, half_t* __restrict__ kmat) {
  long i = blockIdx.x * (long)blockDim.x + threadIdx.x;
  long nv = (long)NIMG * (VPAD - KP) * ED;
  long nq = (long)NIMG * (QPAD - KP) * DQK;
  long nk = (long)NIMG * QPAD * (VPAD - QPAD);
  if (i < nv) {
    long n = i / ((VPAD - KP) * ED), r = i % ((VPAD - KP) * ED);
    vpn[(n * VPAD + KP) * ED + r] = (half_t)0.f;
    return;
  }
  i -= nv;
  if (i < nq) {
    long n = i / ((QPAD - KP) * DQK), r = i % ((QPAD - KP) * DQK);
    qpn[(n * QPAD + KP) * DQK + r] = (half_t)0.f;
    kpn[(n * QPAD + KP) * DQK + r] = (half_t)0.f;
    return;
  }
  i -= nq;
  if (i < nk) {
    long n = i / (QPAD * (VPAD - QPAD)), r = i % (QPAD * (VPAD - QPAD));
    long mrow = r / (VPAD - QPAD), cc = r % (VPAD - QPAD);
    kmat[(n * QPAD + mrow) * VPAD + QPAD + cc] = (half_t)0.f;
  }
}

// ---------------- QK: per-n q(144x128) @ k^T -> relu(qk/sqrtS)^2 -------------

__global__ void k_qk(const half_t* __restrict__ qpn, const half_t* __restrict__ kpn,
                     half_t* __restrict__ kmat) {
  int lane = threadIdx.x;
  int n = blockIdx.z;
  const half_t* Q = qpn + (long)n * QPAD * DQK;
  const half_t* Kb = kpn + (long)n * QPAD * DQK;
  int m0 = blockIdx.y * 16, n0 = blockIdx.x * 16;
  v8f acc = gemm_bT<4>(Q, DQK, Kb, DQK, m0, n0, lane);
  int h = lane >> 4, c = n0 + (lane & 15);
  const float inv_sqrt_s = 0.08838834764831845f;  // 1/sqrt(128)
#pragma unroll
  for (int v = 0; v < 8; ++v) {
    int m = m0 + v + 8 * h;
    float r = fmaxf(acc[v] * inv_sqrt_s, 0.f);
    kmat[((long)n * QPAD + m) * VPAD + c] = (half_t)(r * r);
  }
}

// ---------------- attn = u * (kernel(144x160) @ v(160x512)) ------------------

__global__ void k_attnv(const half_t* __restrict__ kmat, const half_t* __restrict__ vpn,
                        const half_t* __restrict__ u, half_t* __restrict__ attn) {
  int lane = threadIdx.x & 31, wave = threadIdx.x >> 5;
  int n = blockIdx.z;
  int m0 = blockIdx.y * 16, n0 = (blockIdx.x * 4 + wave) * 16;
  const half_t* A = kmat + (long)n * QPAD * VPAD;
  const half_t* B = vpn + (long)n * VPAD * ED;
  v8f acc = gemm_brm<5>(A, VPAD, B, ED, m0, n0, lane);
  int h = lane >> 4, c = n0 + (lane & 15);
#pragma unroll
  for (int v = 0; v < 8; ++v) {
    int m = m0 + v + 8 * h;
    if (m < KP) {
      long rowg = (long)n * KP + m;
      float uval = (float)u[rowg * ED + c];
      attn[rowg * ED + c] = (half_t)(uval * acc[v]);
    }
  }
}

// ---------------- out-proj + residual: feats*res_scale + attn @ o_w^T --------

__global__ void k_gemm_o(const half_t* __restrict__ A, const half_t* __restrict__ W,
                         const float* __restrict__ feats, const float* __restrict__ rs,
                         half_t* __restrict__ f2) {
  int lane = threadIdx.x & 31, wave = threadIdx.x >> 5;
  int m0 = blockIdx.y * 16, n0 = (blockIdx.x * 4 + wave) * 16;
  v8f acc = gemm_bT<16>(A, ED, W, ED, m0, n0, lane);
  int h = lane >> 4, c = n0 + (lane & 15);
#pragma unroll
  for (int v = 0; v < 8; ++v) {
    long row = m0 + v + 8 * h;
    float val = feats[row * DM + c] * rs[c] + acc[v];
    f2[row * DM + c] = (half_t)val;
  }
}

// ---------------- classifiers + flip-blend epilogue --------------------------
// mode 0: bx = 0.5*px      mode 1: bx[flip(kk)][Nout-1-c] += 0.5*px
// mode 2: by[flip(kk)][c] += 0.5*py
__global__ void k_cls(const half_t* __restrict__ A, const half_t* __restrict__ W,
                      float* __restrict__ out, int Nout, int mode) {
  int lane = threadIdx.x & 31, wave = threadIdx.x >> 5;
  int m0 = blockIdx.y * 16, n0 = (blockIdx.x * 4 + wave) * 16;
  v8f acc = gemm_bT<8>(A, DM, W, DM, m0, n0, lane);
  int h = lane >> 4, c = n0 + (lane & 15);
#pragma unroll
  for (int v = 0; v < 8; ++v) {
    int row = m0 + v + 8 * h;
    float val = 0.5f * acc[v];
    if (mode == 0) {
      out[(long)row * Nout + c] = val;
    } else {
      int nimg = row / KP, kk = row % KP;
      int fk = c_FLIP[kk];
      int cc = (mode == 1) ? (Nout - 1 - c) : c;
      out[((long)nimg * KP + fk) * Nout + cc] += val;  // unique dest per src
    }
  }
}

// ---------------- final max/argmax/min reduction -----------------------------

__global__ void k_reduce(const float* __restrict__ bx, const float* __restrict__ by,
                         float* __restrict__ out) {
  int row = blockIdx.x * blockDim.x + threadIdx.x;
  if (row >= ROWS) return;
  const float* px = bx + (long)row * NX;
  float xv = px[0]; int xl = 0;
  for (int j = 1; j < NX; ++j) if (px[j] > xv) { xv = px[j]; xl = j; }
  const float* py = by + (long)row * NY;
  float yv = py[0]; int yl = 0;
  for (int j = 1; j < NY; ++j) if (py[j] > yv) { yv = py[j]; yl = j; }
  float vals = fminf(xv, yv);
  float kx = (vals <= 0.f) ? -1.f : (float)xl;
  float ky = (vals <= 0.f) ? -1.f : (float)yl;
  out[(long)row * 2 + 0] = kx * 0.5f;
  out[(long)row * 2 + 1] = ky * 0.5f;
  out[2L * ROWS + row] = vals;
}

// ---------------- host: workspace carve + launch sequence --------------------

static inline int blks(long n, int t) { return (int)((n + t - 1) / t); }

extern "C" void kernel_launch(void* const* d_in, const int* in_sizes, int n_in,
                              void* d_out, int out_size, void* d_ws, size_t ws_size,
                              hipStream_t stream) {
  const float* f1   = (const float*)d_in[0];
  const float* f2   = (const float*)d_in[1];
  const float* conv_w = (const float*)d_in[2];
  const float* conv_b = (const float*)d_in[3];
  const float* mlp_g  = (const float*)d_in[4];
  const float* mlp_w  = (const float*)d_in[5];
  const float* ln_g   = (const float*)d_in[6];
  const float* uv_w   = (const float*)d_in[7];
  const float* gamma  = (const float*)d_in[8];
  const float* beta   = (const float*)d_in[9];
  const float* o_w    = (const float*)d_in[10];
  const float* res_scale = (const float*)d_in[11];
  const float* clsx_w = (const float*)d_in[12];
  const float* clsy_w = (const float*)d_in[13];

  char* w = (char*)d_ws;
  auto alloc = [&](size_t bytes) {
    char* p = w;
    w += (bytes + 255) & ~(size_t)255;
    return p;
  };
  half_t* convW = (half_t*)alloc((size_t)MPADC * KCONV * 2);
  half_t* mlpW  = (half_t*)alloc((size_t)DM * 128 * 2);
  half_t* uvW   = (half_t*)alloc((size_t)1152 * DM * 2);
  half_t* oW    = (half_t*)alloc((size_t)DM * ED * 2);
  half_t* cxW   = (half_t*)alloc((size_t)NX * DM * 2);
  half_t* cyW   = (half_t*)alloc((size_t)NY * DM * 2);
  half_t* finH  = (half_t*)alloc((size_t)NIMG * CIN * HW * 2);
  float*  feats0 = (float*)alloc((size_t)ROWS * HW * 4);
  half_t* xA    = (half_t*)alloc((size_t)ROWS * 128 * 2);
  float*  featsF = (float*)alloc((size_t)ROWS * DM * 4);
  half_t* xln   = (half_t*)alloc((size_t)ROWS * DM * 2);
  half_t* uH    = (half_t*)alloc((size_t)ROWS * ED * 2);
  half_t* vPN   = (half_t*)alloc((size_t)NIMG * VPAD * ED * 2);
  half_t* qPN   = (half_t*)alloc((size_t)NIMG * QPAD * DQK * 2);
  half_t* kPN   = (half_t*)alloc((size_t)NIMG * QPAD * DQK * 2);
  half_t* kmat  = (half_t*)alloc((size_t)NIMG * QPAD * VPAD * 2);
  half_t* attnH = (half_t*)alloc((size_t)ROWS * ED * 2);
  half_t* f2H   = (half_t*)alloc((size_t)ROWS * DM * 2);
  float*  bx    = (float*)alloc((size_t)ROWS * NX * 4);
  float*  by    = (float*)alloc((size_t)ROWS * NY * 4);

  // weight prepack -> f16 (once per launch; deterministic)
  k_cvt_pad<<<blks((long)MPADC * KCONV, 256), 256, 0, stream>>>(convW, conv_w, MPADC, KP, KCONV, KCONV);
  k_cvt_pad<<<blks((long)DM * 128, 256), 256, 0, stream>>>(mlpW, mlp_w, DM, DM, 128, 108);
  k_cvt_pad<<<blks((long)1152 * DM, 256), 256, 0, stream>>>(uvW, uv_w, 1152, 1152, DM, DM);
  k_cvt_pad<<<blks((long)DM * ED, 256), 256, 0, stream>>>(oW, o_w, DM, DM, ED, ED);
  k_cvt_pad<<<blks((long)NX * DM, 256), 256, 0, stream>>>(cxW, clsx_w, NX, NX, DM, DM);
  k_cvt_pad<<<blks((long)NY * DM, 256), 256, 0, stream>>>(cyW, clsy_w, NY, NY, DM, DM);

  long padTotal = (long)NIMG * (VPAD - KP) * ED + (long)NIMG * (QPAD - KP) * DQK +
                  (long)NIMG * QPAD * (VPAD - QPAD);

  for (int br = 0; br < 2; ++br) {
    const float* f = br ? f2 : f1;
    // input -> f16
    k_cvt_pad<<<blks((long)NIMG * CIN * HW, 256), 256, 0, stream>>>(
        finH, f, NIMG * CIN, NIMG * CIN, HW, HW);
    // conv as implicit GEMM: TDM-staged A tiles + LDS-staged im2col B tiles
    k_conv<<<dim3(NPOS / 16, MPADC / 48), 96, 0, stream>>>(convW, finH, conv_b, feats0);
    // scale_norm(108) -> f16 padded K=128
    k_norm<<<blks(ROWS, 128), 128, 0, stream>>>(feats0, xA, HW, 128, mlp_g);
    // mlp GEMM -> feats f32
    k_gemm_mlp<<<dim3(4, ROWS / 16), 128, 0, stream>>>(xA, mlpW, featsF);
    // scale_norm(256) -> f16
    k_norm<<<blks(ROWS, 128), 128, 0, stream>>>(featsF, xln, DM, DM, ln_g);
    // uv GEMM + SiLU + split
    k_gemm_uv<<<dim3(18, ROWS / 16), 128, 0, stream>>>(xln, uvW, uH, vPN, qPN, kPN, gamma, beta);
    // zero attention padding
    k_padzero<<<blks(padTotal, 256), 256, 0, stream>>>(vPN, qPN, kPN, kmat);
    // qk + relu^2 kernel
    k_qk<<<dim3(QPAD / 16, QPAD / 16, NIMG), 32, 0, stream>>>(qPN, kPN, kmat);
    // kernel @ v, * u
    k_attnv<<<dim3(ED / 64, QPAD / 16, NIMG), 128, 0, stream>>>(kmat, vPN, uH, attnH);
    // out proj + residual
    k_gemm_o<<<dim3(4, ROWS / 16), 128, 0, stream>>>(attnH, oW, featsF, res_scale, f2H);
    // classifiers with flip-blend epilogue
    k_cls<<<dim3(NX / 64, ROWS / 16), 128, 0, stream>>>(f2H, cxW, bx, NX, br ? 1 : 0);
    k_cls<<<dim3(NY / 64, ROWS / 16), 128, 0, stream>>>(f2H, cyW, by, NY, br ? 2 : 0);
  }

  k_reduce<<<blks(ROWS, 128), 128, 0, stream>>>(bx, by, (float*)d_out);
}
